// demAEndGANTimeGenerator_75849122448206
// MI455X (gfx1250) — compile-verified, matrix operand order checked by reading, not run on hardware
//
#include <hip/hip_runtime.h>
#include <hip/hip_bf16.h>
#include <stdint.h>

// ============================================================================
// DoppelGANger generator for MI455X (gfx1250, wave32, WMMA).
//   - bf16 WMMA (v_wmma_f32_16x16x32_bf16) for all GEMMs, f32 accum.
//   - persistent LSTM kernel (64 blocks x 256 thr) with global sense barrier:
//     the 256-step recurrence is latency bound; per-step GEMMs are WMMA work.
//   - each block owns 2 of 128 N-tiles of the gate GEMMs; its ~106KB weight
//     slice is staged to LDS ONCE and feeds WMMA B-operands (ds_load_b128)
//     for all 256 steps -> no per-step weight traffic from L2.
//   - one A fragment feeds 2 WMMAs (dual accumulators, A reuse).
// ============================================================================

typedef __bf16 bf16;
typedef __attribute__((ext_vector_type(16))) __bf16 v16bf;
typedef __attribute__((ext_vector_type(8)))  float   v8f;

#define BB   512
#define TT   256
#define HH   512
#define G4H  2048
#define LDX  160      // Xin cols: [0,15) attr_disc, [15,47) z_t, [47,135) fb, [135,160) zero pad
#define NBLK 64       // LSTM grid: 64 blocks x 2 owned n-tiles = 128 n-tiles

// LDS-resident weight slices (per block): layer0 21kt x 2nt, layer1 32kt x 2nt
#define LW0_FRAGS (21 * 2)
#define LW1_FRAGS (32 * 2)
#define LW0_ELEMS (LW0_FRAGS * 512)
#define LW1_ELEMS (LW1_FRAGS * 512)
#define LSTM_LDS_BYTES ((LW0_ELEMS + LW1_ELEMS) * 2)   // 108544 B < 320KB WGP LDS

// ---------------- workspace layout (bytes) ----------------
#define OFF_WT0   0ull
#define SZ_WT0    (21ull*128*512*2)        // packed [K=672 x N=2048] bf16
#define OFF_WT1   (OFF_WT0 + SZ_WT0)
#define SZ_WT1    (32ull*128*512*2)        // packed [K=1024 x N=2048]
#define OFF_WHD   (OFF_WT1 + SZ_WT1)
#define SZ_WHD    (16ull*6*512*2)          // packed [K=512 x N=96] head weights
#define OFF_AW0   (OFF_WHD + SZ_WHD)
#define SZ_AW0    (1ull*32*512*2)          // packed [32 x 512]
#define OFF_AW1   (OFF_AW0 + SZ_AW0)
#define SZ_AW1    (16ull*32*512*2)         // packed [512 x 512]
#define OFF_AWO   (OFF_AW1 + SZ_AW1)
#define SZ_AWO    (16ull*1*512*2)          // packed [512 x 16]
#define OFF_NOISE (OFF_AWO + SZ_AWO)
#define SZ_NOISE  (512ull*32*2)            // attr noise bf16
#define OFF_XIN   (OFF_NOISE + SZ_NOISE)
#define SZ_XIN    (512ull*160*2)           // LSTM step input, bf16
#define OFF_HBF0  (OFF_XIN + SZ_XIN)
#define SZ_H      (512ull*512*2)
#define OFF_HBF1  (OFF_HBF0 + SZ_H)
#define OFF_A0BF  (OFF_HBF1 + SZ_H)
#define OFF_A1BF  (OFF_A0BF + SZ_H)
#define OFF_A0F   (OFF_A1BF + SZ_H)
#define SZ_AF     (512ull*512*4)
#define OFF_A1F   (OFF_A0F + SZ_AF)
#define OFF_C0S   (OFF_A1F + SZ_AF)
#define OFF_C1S   (OFF_C0S + SZ_AF)
#define OFF_G0    (OFF_C1S + SZ_AF)
#define SZ_G      (512ull*2048*4)
#define OFF_G1    (OFF_G0 + SZ_G)
#define OFF_B0    (OFF_G1 + SZ_G)
#define OFF_B1    (OFF_B0 + 2048ull*4)
#define OFF_BHD   (OFF_B1 + 2048ull*4)
#define OFF_LOG   (OFF_BHD + 128ull*4)
#define OFF_SYNC  (OFF_LOG + 512ull*16*4)

// ---------------- helpers ----------------
__device__ __forceinline__ bf16 f2bf(float f) {
  unsigned u = __float_as_uint(f);
  unsigned r = (u + 0x7FFFu + ((u >> 16) & 1u)) >> 16;
  unsigned short s = (unsigned short)r;
  return __builtin_bit_cast(bf16, s);
}
__device__ __forceinline__ float sigf(float x) { return 1.f / (1.f + __expf(-x)); }

// WMMA wrapper: D = A(16x32 bf16) * B(32x16 bf16) + C(16x16 f32)
__device__ __forceinline__ v8f wmma_bf16(v16bf a, v16bf b, v8f c) {
  return __builtin_amdgcn_wmma_f32_16x16x32_bf16(false, a, false, b, (short)0, c, false, false);
}

// A fragment from row-major [M x K] bf16, tile origin (m0, k0).
// ISA 16-bit A layout: lane group g: elems 0..7 -> K=8g+j, elems 8..15 -> K=16+8g+(j-8)
__device__ __forceinline__ v16bf load_a_frag(const bf16* __restrict__ A, int lda, int m0, int k0) {
  int lane = threadIdx.x & 31;
  int g = lane >> 4, m = lane & 15;
  const bf16* p = A + (size_t)(m0 + m) * lda + k0 + 8 * g;
  union { uint4 q[2]; v16bf v; } u;
  u.q[0] = *(const uint4*)(p);
  u.q[1] = *(const uint4*)(p + 16);
  return u.v;
}

// B fragment from pre-packed fragment-major weights in GLOBAL: one 32B load per lane.
__device__ __forceinline__ v16bf load_b_pk(const bf16* __restrict__ Wpk, int kt, int nt, int NT) {
  int lane = threadIdx.x & 31;
  const v16bf* p = (const v16bf*)Wpk;
  return p[((size_t)kt * NT + nt) * 32 + lane];
}

// B fragment from LDS-resident fragment-major slice.
__device__ __forceinline__ v16bf load_b_lds(const bf16* L, int frag) {
  int lane = threadIdx.x & 31;
  const v16bf* p = (const v16bf*)L;
  return p[frag * 32 + lane];
}

// global sense barrier (counter + generation), s_sleep in spin
__device__ __forceinline__ void gbar(unsigned* sync, unsigned nb) {
  __syncthreads();
  if (threadIdx.x == 0) {
    __threadfence();
    unsigned* cnt = sync;
    unsigned* gen = sync + 1;
    unsigned g = __hip_atomic_load(gen, __ATOMIC_ACQUIRE, __HIP_MEMORY_SCOPE_AGENT);
    unsigned prev = __hip_atomic_fetch_add(cnt, 1u, __ATOMIC_ACQ_REL, __HIP_MEMORY_SCOPE_AGENT);
    if (prev == nb - 1u) {
      __hip_atomic_store(cnt, 0u, __ATOMIC_RELAXED, __HIP_MEMORY_SCOPE_AGENT);
      __hip_atomic_fetch_add(gen, 1u, __ATOMIC_ACQ_REL, __HIP_MEMORY_SCOPE_AGENT);
    } else {
      while (__hip_atomic_load(gen, __ATOMIC_ACQUIRE, __HIP_MEMORY_SCOPE_AGENT) == g)
        __builtin_amdgcn_s_sleep(2);
    }
    __threadfence();
  }
  __syncthreads();
}

// ---------------- weight pre-pack (fp32 -> bf16, fragment-major) ----------------
__global__ void pack_b_kernel(char* ws, int mode, const float* __restrict__ w0,
                              const float* __restrict__ w1) {
  bf16* dst; int KT, NT;
  switch (mode) {
    case 0:  dst = (bf16*)(ws + OFF_WT0); KT = 21; NT = 128; break;
    case 1:  dst = (bf16*)(ws + OFF_WT1); KT = 32; NT = 128; break;
    case 2:  dst = (bf16*)(ws + OFF_WHD); KT = 16; NT = 6;   break;
    case 3:  dst = (bf16*)(ws + OFF_AW0); KT = 1;  NT = 32;  break;
    case 4:  dst = (bf16*)(ws + OFF_AW1); KT = 16; NT = 32;  break;
    default: dst = (bf16*)(ws + OFF_AWO); KT = 16; NT = 1;   break;
  }
  long total = (long)KT * NT * 512;
  long stride = (long)gridDim.x * blockDim.x;
  for (long e = (long)blockIdx.x * blockDim.x + threadIdx.x; e < total; e += stride) {
    int j = (int)(e & 15);
    int lane = (int)((e >> 4) & 31);
    long tile = e >> 9;
    int nt = (int)(tile % NT), kt = (int)(tile / NT);
    int g = lane >> 4;
    int k = kt * 32 + ((j < 8) ? (8 * g + j) : (8 + 8 * g + j));
    int col = nt * 16 + (lane & 15);
    float v = 0.f;
    switch (mode) {
      case 0:  // [W_ih0 | pad | W_hh0]^T : rows 0..134 ih (K=135), 135..159 zero, 160..671 hh
        if (k < 135) v = w0[col * 135 + k];
        else if (k >= 160) v = w1[col * 512 + (k - 160)];
        break;
      case 1:  // [W_ih1 | W_hh1]^T
        v = (k < 512) ? w0[col * 512 + k] : w1[col * 512 + (k - 512)];
        break;
      case 2: { // head: cols 0..71 cont (s*9+d), 72..87 flag (s*2+d), 88..95 pad
        if (col < 72) { int s = col / 9, d = col - s * 9; v = w0[(s * 512 + k) * 9 + d]; }
        else if (col < 88) { int c = col - 72; int s = c >> 1, d = c & 1; v = w1[(s * 512 + k) * 2 + d]; }
        break; }
      case 3:  v = (k < 32) ? w0[k * 512 + col] : 0.f; break;   // attr_W0 [32x512]
      case 4:  v = w0[k * 512 + col]; break;                    // attr_W1 [512x512]
      default: // attr out: cols 0..9 Wd, 10..14 Wc, 15 pad
        if (col < 10) v = w0[k * 10 + col];
        else if (col < 15) v = w1[k * 5 + (col - 10)];
        break;
    }
    dst[e] = f2bf(v);
  }
}

__global__ void prep_misc_kernel(char* ws,
    const float* __restrict__ bih0, const float* __restrict__ bhh0,
    const float* __restrict__ bih1, const float* __restrict__ bhh1,
    const float* __restrict__ out_bc, const float* __restrict__ out_bd,
    const float* __restrict__ attr_noise) {
  float* b0  = (float*)(ws + OFF_B0);
  float* b1  = (float*)(ws + OFF_B1);
  float* bhd = (float*)(ws + OFF_BHD);
  bf16*  nz  = (bf16*)(ws + OFF_NOISE);
  unsigned* sync = (unsigned*)(ws + OFF_SYNC);
  int i = blockIdx.x * blockDim.x + threadIdx.x;
  int n = gridDim.x * blockDim.x;
  for (int e = i; e < 2048; e += n) b0[e] = bih0[e] + bhh0[e];
  for (int e = i; e < 2048; e += n) b1[e] = bih1[e] + bhh1[e];
  for (int e = i; e < 128; e += n) {
    float v = 0.f;
    if (e < 72) { int s = e / 9, d = e - s * 9; v = out_bc[s * 9 + d]; }
    else if (e < 88) { int c = e - 72; int s = c >> 1, d = c & 1; v = out_bd[s * 2 + d]; }
    bhd[e] = v;
  }
  for (int e = i; e < 512 * 32; e += n) nz[e] = f2bf(attr_noise[e]);
  for (int e = i; e < 16; e += n) sync[e] = 0u;
}

// ---------------- attribute MLP (single workgroup: batchnorm needs whole batch) ----------------
__global__ void __launch_bounds__(512) attr_kernel(char* ws,
    const float* __restrict__ ab0, const float* __restrict__ ag0, const float* __restrict__ abe0,
    const float* __restrict__ ab1, const float* __restrict__ ag1, const float* __restrict__ abe1,
    const float* __restrict__ bd, const float* __restrict__ bc,
    float* __restrict__ outAttr) {
  const bf16* nz  = (const bf16*)(ws + OFF_NOISE);
  const bf16* aw0 = (const bf16*)(ws + OFF_AW0);
  const bf16* aw1 = (const bf16*)(ws + OFF_AW1);
  const bf16* awo = (const bf16*)(ws + OFF_AWO);
  bf16* a0bf = (bf16*)(ws + OFF_A0BF);
  bf16* a1bf = (bf16*)(ws + OFF_A1BF);
  bf16* xin  = (bf16*)(ws + OFF_XIN);
  float* a0f = (float*)(ws + OFF_A0F);
  float* a1f = (float*)(ws + OFF_A1F);
  float* logits = (float*)(ws + OFF_LOG);
  const int wave = threadIdx.x >> 5;
  const int lane = threadIdx.x & 31;
  const int g = lane >> 4, nc = lane & 15;

  // layer0: relu(noise @ W0 + b0)
  for (int tile = wave; tile < 32 * 32; tile += 16) {
    int mt = tile >> 5, nt = tile & 31;
    int m0 = mt * 16, n0 = nt * 16;
    v8f acc = {};
    v16bf a = load_a_frag(nz, 32, m0, 0);
    v16bf b = load_b_pk(aw0, 0, nt, 32);
    acc = wmma_bf16(a, b, acc);
    float bias = ab0[n0 + nc];
#pragma unroll
    for (int r = 0; r < 8; ++r) {
      int row = m0 + r + 8 * g;
      a0f[(size_t)row * 512 + n0 + nc] = fmaxf(acc[r] + bias, 0.f);
    }
  }
  __syncthreads();
  // batchnorm 0 (train mode, biased var) -> bf16
  {
    int c = threadIdx.x;
    float m = 0.f, m2 = 0.f;
    for (int b = 0; b < 512; ++b) { float x = a0f[(size_t)b * 512 + c]; m += x; m2 += x * x; }
    m *= (1.f / 512.f); float v = m2 * (1.f / 512.f) - m * m;
    float sc = ag0[c] * rsqrtf(v + 1e-5f);
    float sh = abe0[c] - m * sc;
    for (int b = 0; b < 512; ++b)
      a0bf[(size_t)b * 512 + c] = f2bf(a0f[(size_t)b * 512 + c] * sc + sh);
  }
  __syncthreads();
  // layer1
  for (int tile = wave; tile < 32 * 32; tile += 16) {
    int mt = tile >> 5, nt = tile & 31;
    int m0 = mt * 16, n0 = nt * 16;
    v8f acc = {};
    for (int kt = 0; kt < 16; ++kt) {
      v16bf a = load_a_frag(a0bf, 512, m0, kt * 32);
      v16bf b = load_b_pk(aw1, kt, nt, 32);
      acc = wmma_bf16(a, b, acc);
    }
    float bias = ab1[n0 + nc];
#pragma unroll
    for (int r = 0; r < 8; ++r) {
      int row = m0 + r + 8 * g;
      a1f[(size_t)row * 512 + n0 + nc] = fmaxf(acc[r] + bias, 0.f);
    }
  }
  __syncthreads();
  // batchnorm 1
  {
    int c = threadIdx.x;
    float m = 0.f, m2 = 0.f;
    for (int b = 0; b < 512; ++b) { float x = a1f[(size_t)b * 512 + c]; m += x; m2 += x * x; }
    m *= (1.f / 512.f); float v = m2 * (1.f / 512.f) - m * m;
    float sc = ag1[c] * rsqrtf(v + 1e-5f);
    float sh = abe1[c] - m * sc;
    for (int b = 0; b < 512; ++b)
      a1bf[(size_t)b * 512 + c] = f2bf(a1f[(size_t)b * 512 + c] * sc + sh);
  }
  __syncthreads();
  // output heads GEMM -> logits [512 x 16]
  for (int tile = wave; tile < 32; tile += 16) {
    int m0 = tile * 16;
    v8f acc = {};
    for (int kt = 0; kt < 16; ++kt) {
      v16bf a = load_a_frag(a1bf, 512, m0, kt * 32);
      v16bf b = load_b_pk(awo, kt, 0, 1);
      acc = wmma_bf16(a, b, acc);
    }
#pragma unroll
    for (int r = 0; r < 8; ++r)
      logits[(size_t)(m0 + r + 8 * g) * 16 + nc] = acc[r];
  }
  __syncthreads();
  // per-row: softmax(10) + argmax one-hot, sigmoid(5); fill Xin attr cols
  {
    int b = threadIdx.x;
    float l[10]; float mx = -1e30f; int am = 0;
    for (int c = 0; c < 10; ++c) {
      float x = logits[(size_t)b * 16 + c] + bd[c];
      l[c] = x;
      if (x > mx) { mx = x; am = c; }
    }
    float se = 0.f;
    for (int c = 0; c < 10; ++c) { l[c] = __expf(l[c] - mx); se += l[c]; }
    float inv = 1.f / se;
    for (int c = 0; c < 10; ++c) outAttr[(size_t)b * 15 + c] = l[c] * inv;
    for (int c = 0; c < 5; ++c) {
      float s = sigf(logits[(size_t)b * 16 + 10 + c] + bc[c]);
      outAttr[(size_t)b * 15 + 10 + c] = s;
      xin[(size_t)b * LDX + 10 + c] = f2bf(s);
    }
    for (int c = 0; c < 10; ++c)
      xin[(size_t)b * LDX + c] = f2bf(c == am ? 1.f : 0.f);
  }
}

// ---------------- persistent LSTM + heads kernel ----------------
// Block b owns gate-GEMM n-tiles {2b, 2b+1}; their weight slice lives in LDS.
__global__ void __launch_bounds__(256) lstm_kernel(char* ws,
    const float* __restrict__ fnoise, const float* __restrict__ fdata,
    const float* __restrict__ h0in, const float* __restrict__ c0in,
    float* __restrict__ outFeat) {
  extern __shared__ __align__(16) bf16 smem[];
  bf16* lw0 = smem;               // 21kt x 2nt fragments
  bf16* lw1 = smem + LW0_ELEMS;   // 32kt x 2nt fragments

  bf16* xin  = (bf16*)(ws + OFF_XIN);
  bf16* hbf0 = (bf16*)(ws + OFF_HBF0);
  bf16* hbf1 = (bf16*)(ws + OFF_HBF1);
  const bf16* wt0 = (const bf16*)(ws + OFF_WT0);
  const bf16* wt1 = (const bf16*)(ws + OFF_WT1);
  const bf16* whd = (const bf16*)(ws + OFF_WHD);
  float* c0s = (float*)(ws + OFF_C0S);
  float* c1s = (float*)(ws + OFF_C1S);
  float* g0b = (float*)(ws + OFF_G0);
  float* g1b = (float*)(ws + OFF_G1);
  const float* b0  = (const float*)(ws + OFF_B0);
  const float* b1  = (const float*)(ws + OFF_B1);
  const float* bhd = (const float*)(ws + OFF_BHD);
  unsigned* sync = (unsigned*)(ws + OFF_SYNC);

  const int gtid = blockIdx.x * 256 + threadIdx.x;
  const int gsz  = gridDim.x * 256;
  const int wave = gtid >> 5;
  const int wib  = threadIdx.x >> 5;         // wave in block, 0..7
  const int lane = threadIdx.x & 31;
  const int g = lane >> 4, nc = lane & 15;
  const unsigned nb = gridDim.x;
  const int ntBase = 2 * blockIdx.x;         // owned n-tiles: ntBase, ntBase+1
  const int n0blk = ntBase * 16;

  // ---- stage owned weight slices into LDS (once; reused for all 256 steps) ----
  {
    const uint4* src0 = (const uint4*)wt0;
    const uint4* src1 = (const uint4*)wt1;
    uint4* d0 = (uint4*)lw0;
    uint4* d1 = (uint4*)lw1;
    // each fragment = 512 bf16 = 64 uint4
    for (int i = threadIdx.x; i < LW0_FRAGS * 64; i += 256) {
      int frag = i >> 6, rem = i & 63;
      int kt = frag >> 1, ntl = frag & 1;
      d0[i] = src0[((size_t)(kt * 128 + ntBase + ntl) * 512) / 8 + rem];
    }
    for (int i = threadIdx.x; i < LW1_FRAGS * 64; i += 256) {
      int frag = i >> 6, rem = i & 63;
      int kt = frag >> 1, ntl = frag & 1;
      d1[i] = src1[((size_t)(kt * 128 + ntBase + ntl) * 512) / 8 + rem];
    }
  }

  // ---- init: states fp32->bf16/fp32, feedback cols, zero pad cols ----
  for (int i = gtid; i < 512 * 512; i += gsz) {
    hbf0[i] = f2bf(h0in[i]);
    hbf1[i] = f2bf(h0in[512 * 512 + i]);
    c0s[i] = c0in[i];
    c1s[i] = c0in[512 * 512 + i];
  }
  for (int i = gtid; i < 512 * 88; i += gsz) {
    int b = i / 88, c = i - b * 88;
    xin[(size_t)b * LDX + 47 + c] = f2bf(fdata[i]);
  }
  for (int i = gtid; i < 512 * 25; i += gsz) {
    int b = i / 25, c = i - b * 25;
    xin[(size_t)b * LDX + 135 + c] = f2bf(0.f);
  }
  gbar(sync, nb);   // includes __syncthreads -> LDS staging visible

  for (int t = 0; t < TT; ++t) {
    // ---- phase A: write z_t into Xin ----
    for (int i = gtid; i < 512 * 32; i += gsz) {
      int b = i >> 5, c = i & 31;
      xin[(size_t)b * LDX + 15 + c] = f2bf(fnoise[(size_t)b * (TT * 32) + (size_t)t * 32 + c]);
    }
    gbar(sync, nb);

    // ---- phase B: gates0 = [Xin | h0] @ Wt0 (B from LDS, dual n accumulators) ----
    for (int mt = wib; mt < 32; mt += 8) {
      int m0 = mt * 16;
      v8f acc0 = {}, acc1 = {};
      for (int kt = 0; kt < 5; ++kt) {
        v16bf a = load_a_frag(xin, LDX, m0, kt * 32);
        acc0 = wmma_bf16(a, load_b_lds(lw0, kt * 2 + 0), acc0);
        acc1 = wmma_bf16(a, load_b_lds(lw0, kt * 2 + 1), acc1);
      }
      for (int kt = 0; kt < 16; ++kt) {
        v16bf a = load_a_frag(hbf0, 512, m0, kt * 32);
        acc0 = wmma_bf16(a, load_b_lds(lw0, (5 + kt) * 2 + 0), acc0);
        acc1 = wmma_bf16(a, load_b_lds(lw0, (5 + kt) * 2 + 1), acc1);
      }
      float* dst = g0b + (size_t)(m0 + 8 * g) * G4H + n0blk + nc;
#pragma unroll
      for (int r = 0; r < 8; ++r) {
        dst[(size_t)r * G4H]      = acc0[r];
        dst[(size_t)r * G4H + 16] = acc1[r];
      }
    }
    gbar(sync, nb);

    // ---- phase C: LSTM cell 0 ----
    for (int idx = gtid; idx < 512 * 512; idx += gsz) {
      int b = idx >> 9, n = idx & 511;
      const float* gr = g0b + (size_t)b * G4H;
      float gi = gr[n]          + b0[n];
      float gf = gr[512 + n]    + b0[512 + n];
      float gg = gr[1024 + n]   + b0[1024 + n];
      float go = gr[1536 + n]   + b0[1536 + n];
      float c  = sigf(gf) * c0s[idx] + sigf(gi) * tanhf(gg);
      c0s[idx] = c;
      hbf0[idx] = f2bf(sigf(go) * tanhf(c));
    }
    gbar(sync, nb);

    // ---- phase D: gates1 = [h0n | h1] @ Wt1 (B from LDS) ----
    for (int mt = wib; mt < 32; mt += 8) {
      int m0 = mt * 16;
      v8f acc0 = {}, acc1 = {};
      for (int kt = 0; kt < 16; ++kt) {
        v16bf a = load_a_frag(hbf0, 512, m0, kt * 32);
        acc0 = wmma_bf16(a, load_b_lds(lw1, kt * 2 + 0), acc0);
        acc1 = wmma_bf16(a, load_b_lds(lw1, kt * 2 + 1), acc1);
      }
      for (int kt = 0; kt < 16; ++kt) {
        v16bf a = load_a_frag(hbf1, 512, m0, kt * 32);
        acc0 = wmma_bf16(a, load_b_lds(lw1, (16 + kt) * 2 + 0), acc0);
        acc1 = wmma_bf16(a, load_b_lds(lw1, (16 + kt) * 2 + 1), acc1);
      }
      float* dst = g1b + (size_t)(m0 + 8 * g) * G4H + n0blk + nc;
#pragma unroll
      for (int r = 0; r < 8; ++r) {
        dst[(size_t)r * G4H]      = acc0[r];
        dst[(size_t)r * G4H + 16] = acc1[r];
      }
    }
    gbar(sync, nb);

    // ---- phase E: LSTM cell 1 ----
    for (int idx = gtid; idx < 512 * 512; idx += gsz) {
      int b = idx >> 9, n = idx & 511;
      const float* gr = g1b + (size_t)b * G4H;
      float gi = gr[n]          + b1[n];
      float gf = gr[512 + n]    + b1[512 + n];
      float gg = gr[1024 + n]   + b1[1024 + n];
      float go = gr[1536 + n]   + b1[1536 + n];
      float c  = sigf(gf) * c1s[idx] + sigf(gi) * tanhf(gg);
      c1s[idx] = c;
      hbf1[idx] = f2bf(sigf(go) * tanhf(c));
    }
    gbar(sync, nb);

    // ---- phase F: output heads (sigmoid cont + softmax2 gen-flag), feed back ----
    if (wave < 192) {  // 32 m-tiles x 6 n-tiles, one per wave (wave-uniform)
      int mt = wave / 6, nt = wave - mt * 6;
      int m0 = mt * 16, n0 = nt * 16;
      v8f acc = {};
      for (int kt = 0; kt < 16; ++kt) {
        v16bf a = load_a_frag(hbf1, 512, m0, kt * 32);
        v16bf b = load_b_pk(whd, kt, nt, 6);
        acc = wmma_bf16(a, b, acc);
      }
      int col = n0 + nc;
      float bias = bhd[col];
#pragma unroll
      for (int r = 0; r < 8; ++r) {
        int b = m0 + r + 8 * g;
        float x = acc[r] + bias;
        float xo = __shfl_xor(x, 1, 32);  // softmax pair partner (uniform exec)
        float outv;
        if (col < 72) {
          outv = sigf(x);
        } else {
          float mx = fmaxf(x, xo);
          float ex = __expf(x - mx), eo = __expf(xo - mx);
          outv = ex / (ex + eo);
        }
        if (col < 88) {
          int o;
          if (col < 72) { int s = col / 9; o = s * 11 + (col - s * 9); }
          else { int cc = col - 72; int s = cc >> 1; o = s * 11 + 9 + (cc & 1); }
          outFeat[(size_t)b * (TT * 88) + (size_t)t * 88 + o] = outv;
          xin[(size_t)b * LDX + 47 + o] = f2bf(outv);   // feedback for t+1
        }
      }
    }
    gbar(sync, nb);
  }
}

// ---------------- host entry ----------------
extern "C" void kernel_launch(void* const* d_in, const int* in_sizes, int n_in,
                              void* d_out, int out_size, void* d_ws, size_t ws_size,
                              hipStream_t stream) {
  (void)in_sizes; (void)n_in; (void)out_size; (void)ws_size;
  char* ws = (char*)d_ws;

  // weight packing (fp32 -> bf16, fragment-major)
  pack_b_kernel<<<512, 256, 0, stream>>>(ws, 0, (const float*)d_in[18], (const float*)d_in[19]); // W_ih0, W_hh0
  pack_b_kernel<<<512, 256, 0, stream>>>(ws, 1, (const float*)d_in[22], (const float*)d_in[23]); // W_ih1, W_hh1
  pack_b_kernel<<<64,  256, 0, stream>>>(ws, 2, (const float*)d_in[26], (const float*)d_in[28]); // out_Wc, out_Wd
  pack_b_kernel<<<32,  256, 0, stream>>>(ws, 3, (const float*)d_in[6],  nullptr);                // attr_W0
  pack_b_kernel<<<256, 256, 0, stream>>>(ws, 4, (const float*)d_in[10], nullptr);                // attr_W1
  pack_b_kernel<<<16,  256, 0, stream>>>(ws, 5, (const float*)d_in[14], (const float*)d_in[16]); // attrout_Wd, attrout_Wc

  prep_misc_kernel<<<64, 256, 0, stream>>>(ws,
      (const float*)d_in[20], (const float*)d_in[21],   // b_ih0, b_hh0
      (const float*)d_in[24], (const float*)d_in[25],   // b_ih1, b_hh1
      (const float*)d_in[27], (const float*)d_in[29],   // out_bc, out_bd
      (const float*)d_in[0]);                           // attribute_input_noise

  float* outFeat = (float*)d_out;
  float* outAttr = outFeat + (size_t)512 * 256 * 88;

  attr_kernel<<<1, 512, 0, stream>>>(ws,
      (const float*)d_in[7],  (const float*)d_in[8],  (const float*)d_in[9],   // b0, g0, be0
      (const float*)d_in[11], (const float*)d_in[12], (const float*)d_in[13],  // b1, g1, be1
      (const float*)d_in[15], (const float*)d_in[17],                          // attrout_bd, attrout_bc
      outAttr);

  lstm_kernel<<<NBLK, 256, LSTM_LDS_BYTES, stream>>>(ws,
      (const float*)d_in[2], (const float*)d_in[3],   // feature_input_noise, feature_input_data
      (const float*)d_in[4], (const float*)d_in[5],   // h0, c0
      outFeat);
}
